// IrregularRNN_40956808135367
// MI455X (gfx1250) — matrix operand order repaired and checked by reading, not compiled, not run on hardware
//
#include <hip/hip_runtime.h>
#include <hip/hip_bf16.h>

#if __has_include(<hip/amd_detail/amd_gfx1250_TDM.h>)
#define TDM_6ARG 1
#endif

// Problem constants (from reference): B=128, T=256, D=256, U=1024
static constexpr int Bq = 128;
static constexpr int Tt = 256;
static constexpr int Dd = 256;
static constexpr int Uu = 1024;
static constexpr int Kk = Dd + Uu;        // 1280 combined GEMM K
static constexpr int Nn = 2 * Uu;         // 2048 output columns of pre
static constexpr int NTILES = Nn / 16;    // 128
static constexpr int KCHUNKS = Kk / 32;   // 40
static constexpr int NBLK = 64;           // persistent grid blocks

typedef __attribute__((ext_vector_type(16))) __bf16 v16bf;
typedef __attribute__((ext_vector_type(8)))  float  v8f;
typedef __attribute__((ext_vector_type(4)))  unsigned int u32x4;
typedef __attribute__((ext_vector_type(8)))  int i32x8;
typedef __attribute__((ext_vector_type(4)))  int i32x4;

union BF16x16 { uint4 q[2]; v16bf v; };

__device__ __forceinline__ unsigned short f2bf(float x) {
  unsigned u = __builtin_bit_cast(unsigned, x);
  u += 0x7fffu + ((u >> 16) & 1u);        // round-to-nearest-even
  return (unsigned short)(u >> 16);
}

// TDM pads 4 dwords (16B) after every 128 dwords (512B) stored; map a flat
// bf16 element index E of the tile to its padded LDS byte offset.
// Row = 1280 elems = 2560B -> 5 pads/row -> LDS row stride 2640B = 660 dwords
// (660 mod 64 = 20 banks: 16 consecutive rows land on 16 distinct banks).
__device__ __forceinline__ unsigned ldsOff(int E) {
  return 2u * (unsigned)E + (((unsigned)E >> 8) << 4);
}

// ---------------------------------------------------------------------------
// Issue one TDM 2D tile load: 16 rows x 1280 bf16 from global -> LDS.
// D# group0: count=1(valid), lds_addr, 57b global addr, type=2 ("image").
// D# group1: data_size=2B, pad_enable, pad_interval=128dw, pad_amount=4dw,
//            tensor_dim0=1280, tensor_dim1=128, tile_dim0=1280, tile_dim1=16,
//            tensor_dim0_stride=1280. Groups 2/3 unused (2D tensor).
// ---------------------------------------------------------------------------
__device__ __forceinline__ void tdm_load_slab(const unsigned short* gsrc,
                                              unsigned ldsAddr) {
  unsigned long long ga = (unsigned long long)(uintptr_t)gsrc;
  u32x4 g0 = { 1u,                                   // count=1, is_restore=0
               ldsAddr,                              // lds_addr[31:0]
               (unsigned)(ga & 0xFFFFFFFFu),         // global_addr[31:0]
               (unsigned)(((ga >> 32) & 0x1FFFFFFu) | (2u << 30)) }; // +type=2
  i32x8 g1 = { (int)((1u << 16)                      // data_size = 2 bytes
                   | (1u << 20)                      // pad_enable
                   | (6u << 22)                      // pad_interval: 128 dwords
                   | (3u << 25)),                    // pad_amount: 4 dwords
               (int)((unsigned)Kk << 16),            // tensor_dim0 = 1280 (lo16)
               (int)((unsigned)Bq << 16),            // tensor_dim1 = 128 (lo16)
               (int)((unsigned)Kk << 16),            // tile_dim0 = 1280
               16,                                   // tile_dim1 = 16 rows
               (int)Kk,                              // tensor_dim0_stride[31:0]
               0, 0 };
  i32x4 z4 = { 0, 0, 0, 0 };
#ifdef TDM_6ARG
  i32x8 z8 = { 0, 0, 0, 0, 0, 0, 0, 0 };
  __builtin_amdgcn_tensor_load_to_lds(g0, g1, z4, z4, z8, 0);
#else
  __builtin_amdgcn_tensor_load_to_lds(g0, g1, z4, z4, 0);
#endif
}

// ---------------------------------------------------------------------------
// Pack [Wx; Wh] (f32, row-major [K, 2U]) into bf16 tiles in the exact WMMA
// B-operand lane layout: [nTile][kChunk][lane][16 bf16]; lane 0-15 hold
// K=kc*32+0..7,16..23 (col=lane), lane 16-31 hold K=kc*32+8..15,24..31. 32B/lane coalesced.
// ---------------------------------------------------------------------------
__global__ void k_pack(const float* __restrict__ Wx, const float* __restrict__ Wh,
                       unsigned short* __restrict__ Wc) {
  int idx = blockIdx.x * blockDim.x + threadIdx.x;      // one (nTile,kc,lane)
  if (idx >= NTILES * KCHUNKS * 32) return;
  int lane = idx & 31;
  int kc   = (idx >> 5) % KCHUNKS;
  int nT   = idx / (KCHUNKS * 32);
  int n    = nT * 16 + (lane & 15);
  int kb   = kc * 32 + ((lane >> 4) ? 8 : 0);
  unsigned short* dst = Wc + (size_t)idx * 16;
#pragma unroll
  for (int e = 0; e < 16; ++e) {
    int k = kb + e + ((e >= 8) ? 8 : 0);
    float w = (k < Dd) ? Wx[(size_t)k * Nn + n] : Wh[(size_t)(k - Dd) * Nn + n];
    dst[e] = f2bf(w);
  }
}

// ---------------------------------------------------------------------------
// Per-launch init: A0 = bf16([x_0, h0]), hbuf = h0, tau = softplus(w_tau),
// barrier counters = 0. Re-run every launch -> deterministic / capture-safe.
// ---------------------------------------------------------------------------
__global__ void k_setup(const float* __restrict__ features, const float* __restrict__ h0,
                        const float* __restrict__ w_tau,
                        unsigned short* __restrict__ A0, float* __restrict__ hbuf,
                        float* __restrict__ tau, unsigned* __restrict__ bar) {
  int idx = blockIdx.x * blockDim.x + threadIdx.x;
  if (idx < Bq * Kk) {
    int brow = idx / Kk, col = idx - brow * Kk;
    float v = (col < Dd) ? features[((size_t)brow * Tt) * Dd + col]
                         : h0[(size_t)brow * Uu + (col - Dd)];
    A0[idx] = f2bf(v);
  }
  if (idx < Bq * Uu) hbuf[idx] = h0[idx];
  if (idx < Uu) {
    float w = w_tau[idx];
    tau[idx] = (w > 20.f) ? w : log1pf(__expf(w));
  }
  if (idx < 2) bar[idx] = 0u;
}

// ---------------------------------------------------------------------------
// Persistent LTC-RNN kernel. 64 blocks x 256 threads (8 waves, wave32).
// Block wg: mTile = wg>>3 (16 batch rows); wave owns N-tile pair (p, p+64)
// so each lane gets matched (f, a) pre-activations for the same hidden unit.
// Per step: TDM-DMA the A slab into LDS, 40 K-chunks x 2 WMMA bf16, LTC cell
// math, write h (f32 state + bf16 next-step A operand), stage x_{t+1},
// grid-wide barrier.
// ---------------------------------------------------------------------------
__global__ void __launch_bounds__(256, 1)
k_rnn(const unsigned short* __restrict__ Wc,
      unsigned short* __restrict__ A0, unsigned short* __restrict__ A1,
      float* __restrict__ hbuf, const float* __restrict__ tau,
      const float* __restrict__ features, const float* __restrict__ tsteps,
      const float* __restrict__ bias, float* __restrict__ out,
      unsigned* __restrict__ barCnt, unsigned* __restrict__ barGen) {
  // 16*1280 bf16 + 80 pads * 16B = 42240 B, rounded up
  __shared__ __attribute__((aligned(64))) unsigned char sA[43008];
  __shared__ float sdt[Bq];

  const int wg    = blockIdx.x;
  const int mTile = wg >> 3;
  const int wave  = threadIdx.x >> 5;
  const int lane  = threadIdx.x & 31;
  const int p     = (wg & 7) * 8 + wave;   // 0..63: N-tile pair id
  const int jf = p, ja = p + 64;
  const int r0 = mTile * 16;

  // C/D layout: lane 0-15 -> N=lane, M=vgpr; lane 16-31 -> N=lane-16, M=vgpr+8
  const int n    = lane & 15;
  const int mAdd = (lane < 16) ? 0 : 8;
  const int u    = jf * 16 + n;            // hidden unit index, fixed per lane
  const float bias_f = bias[u];
  const float bias_a = bias[Uu + u];
  const float tauu   = tau[u];
  // A layout: lanes<16 hold K 0..7 & 16..23; lanes>=16 hold K 8..15 & 24..31
  const int kshift = (lane >> 4) ? 8 : 0;
  const int Ebase  = (lane & 15) * Kk;     // lane's A row, flat element index

  const uint4* WcQ = (const uint4*)Wc;
  const unsigned sAaddr = (unsigned)(uintptr_t)&sA[0];

  for (int t = 0; t < Tt; ++t) {
    unsigned short* Acur = (t & 1) ? A1 : A0;
    unsigned short* Anxt = (t & 1) ? A0 : A1;

    // ---- TDM: DMA the 16x1280 bf16 A slab into LDS (wave 0 issues) ----
    if (wave == 0) {
      tdm_load_slab(Acur + (size_t)r0 * Kk, sAaddr);
      __builtin_amdgcn_s_wait_tensorcnt(0);
      __asm__ volatile("" ::: "memory");
    }
    if (threadIdx.x < Bq) sdt[threadIdx.x] = tsteps[(size_t)threadIdx.x * Tt + t];
    __syncthreads();

    // ---- GEMM: pre = [x_t, h] @ [Wx; Wh], bf16 in / f32 accumulate ----
    v8f cf = {};
    v8f ca = {};
#pragma unroll 2
    for (int kc = 0; kc < KCHUNKS; ++kc) {
      BF16x16 af, bf, ba;
      int E0 = Ebase + kc * 32 + kshift;   // K kbase..kbase+7
      int E1 = E0 + 16;                    // K kbase+16..kbase+23
      af.q[0] = *(const uint4*)(sA + ldsOff(E0));
      af.q[1] = *(const uint4*)(sA + ldsOff(E1));
      const uint4* pf = WcQ + ((size_t)(jf * KCHUNKS + kc) * 32 + lane) * 2;
      const uint4* pa = WcQ + ((size_t)(ja * KCHUNKS + kc) * 32 + lane) * 2;
      bf.q[0] = pf[0]; bf.q[1] = pf[1];
      ba.q[0] = pa[0]; ba.q[1] = pa[1];
      __builtin_prefetch((const void*)(pf + 64), 0, 1);  // next kc B tile
      __builtin_prefetch((const void*)(pa + 64), 0, 1);
      cf = __builtin_amdgcn_wmma_f32_16x16x32_bf16(false, af.v, false, bf.v,
                                                   (short)0, cf, false, false);
      ca = __builtin_amdgcn_wmma_f32_16x16x32_bf16(false, af.v, false, ba.v,
                                                   (short)0, ca, false, false);
    }

    // ---- LTC cell: f=sigmoid, a=tanh, h' = (h-a)*exp(-dt*(tau+f)) + a ----
#pragma unroll
    for (int v = 0; v < 8; ++v) {
      int brow = r0 + mAdd + v;
      float dt = sdt[brow];
      float pf = cf[v] + bias_f;
      float pa = ca[v] + bias_a;
      float f  = __builtin_amdgcn_rcpf(1.f + __expf(-pf));          // sigmoid
      float e2 = __expf(2.f * pa);
      float a  = 1.f - 2.f * __builtin_amdgcn_rcpf(e2 + 1.f);       // tanh
      size_t hidx = (size_t)brow * Uu + u;
      float hp = hbuf[hidx];
      float decay = __expf(-dt * (tauu + f));
      float hn = (hp - a) * decay + a;
      hbuf[hidx] = hn;
      out[((size_t)brow * Tt + t) * Uu + u] = hn;
      Anxt[(size_t)brow * Kk + Dd + u] = f2bf(hn);     // h part of next A
    }

    // ---- stage x_{t+1} (bf16) into next A buffer, grid-strided ----
    if (t + 1 < Tt) {
      for (int idx = blockIdx.x * 256 + threadIdx.x; idx < Bq * Dd;
           idx += NBLK * 256) {
        int brow = idx >> 8, dcol = idx & 255;
        Anxt[(size_t)brow * Kk + dcol] =
            f2bf(features[((size_t)brow * Tt + (t + 1)) * Dd + dcol]);
      }
    }

    // ---- grid-wide sense/generation barrier ----
    __syncthreads();
    if (threadIdx.x == 0) {
      __threadfence();
      unsigned g = __hip_atomic_load(barGen, __ATOMIC_ACQUIRE,
                                     __HIP_MEMORY_SCOPE_AGENT);
      if (__hip_atomic_fetch_add(barCnt, 1u, __ATOMIC_ACQ_REL,
                                 __HIP_MEMORY_SCOPE_AGENT) == NBLK - 1) {
        __hip_atomic_store(barCnt, 0u, __ATOMIC_RELAXED,
                           __HIP_MEMORY_SCOPE_AGENT);
        __hip_atomic_fetch_add(barGen, 1u, __ATOMIC_RELEASE,
                               __HIP_MEMORY_SCOPE_AGENT);
      } else {
        while (__hip_atomic_load(barGen, __ATOMIC_ACQUIRE,
                                 __HIP_MEMORY_SCOPE_AGENT) == g)
          __builtin_amdgcn_s_sleep(1);
      }
    }
    __syncthreads();
  }
}

// ---------------------------------------------------------------------------
extern "C" void kernel_launch(void* const* d_in, const int* in_sizes, int n_in,
                              void* d_out, int out_size, void* d_ws, size_t ws_size,
                              hipStream_t stream) {
  const float* features = (const float*)d_in[0];  // [B,T,D]
  const float* tsteps   = (const float*)d_in[1];  // [B,T]
  const float* Wx       = (const float*)d_in[2];  // [D,2U]
  const float* Wh       = (const float*)d_in[3];  // [U,2U]
  const float* bias     = (const float*)d_in[4];  // [2U]
  const float* w_tau    = (const float*)d_in[5];  // [U]
  const float* h0       = (const float*)d_in[6];  // [B,U]
  float* out = (float*)d_out;                     // [B,T,U]

  char* ws = (char*)d_ws;
  const size_t SZ_WC = (size_t)NTILES * KCHUNKS * 32 * 16 * 2;  // 5,242,880
  const size_t SZ_A  = (size_t)Bq * Kk * 2;                     //   327,680
  const size_t SZ_H  = (size_t)Bq * Uu * 4;                     //   524,288
  unsigned short* Wc  = (unsigned short*)(ws);
  unsigned short* A0  = (unsigned short*)(ws + SZ_WC);
  unsigned short* A1  = (unsigned short*)(ws + SZ_WC + SZ_A);
  float*          hbf = (float*)(ws + SZ_WC + 2 * SZ_A);
  float*          tau = (float*)(ws + SZ_WC + 2 * SZ_A + SZ_H);
  unsigned*       bar = (unsigned*)(ws + SZ_WC + 2 * SZ_A + SZ_H + 4096);

  int packN = NTILES * KCHUNKS * 32;
  k_pack<<<(packN + 255) / 256, 256, 0, stream>>>(Wx, Wh, Wc);
  k_setup<<<(Bq * Kk + 255) / 256, 256, 0, stream>>>(features, h0, w_tau,
                                                     A0, hbf, tau, bar);
  k_rnn<<<NBLK, 256, 0, stream>>>(Wc, A0, A1, hbf, tau, features, tsteps,
                                  bias, out, bar, bar + 1);
}
